// Attention_44384192037053
// MI455X (gfx1250) — matrix operand order repaired and checked by reading, not compiled, not run on hardware
//
#include <hip/hip_runtime.h>
#include <hip/hip_bf16.h>

typedef __attribute__((ext_vector_type(16))) _Float16 v16h;
typedef __attribute__((ext_vector_type(8)))  float    v8f;
typedef __attribute__((ext_vector_type(4)))  float    f32x4;

#define B_  16
#define N_  1024
#define C_  1024
#define H_  16
#define DH_ 64
#define M_  (B_ * N_)   // 16384 rows
#define LDA 40          // padded LDS stride (halves) for 32-halve tile rows
#define LDK 72          // padded LDS stride (halves) for 64-halve K rows

// ---------------- WMMA fragment helpers (CDNA5 16x16x32 f16) ----------------
__device__ __forceinline__ v16h load_a(const _Float16* src, int ld, int lane) {
  int row  = lane & 15;
  int half = lane >> 4;
  const _Float16* p = src + (size_t)row * ld + half * 8;
  union { v16h h; struct { f32x4 lo, hi; } q; } u;
  u.q.lo = *(const f32x4*)(p);
  u.q.hi = *(const f32x4*)(p + 16);
  return u.h;
}

__device__ __forceinline__ v16h load_b(const _Float16* src, int ld, int lane) {
  int n  = lane & 15;
  int kh = lane >> 4;
  const _Float16* p = src + (size_t)n * ld + kh * 8;
  union { v16h h; struct { f32x4 lo, hi; } q; } u;
  u.q.lo = *(const f32x4*)(p);
  u.q.hi = *(const f32x4*)(p + 16);
  return u.h;
}

__device__ __forceinline__ v8f wmma16(v16h a, v16h b, v8f c) {
  return __builtin_amdgcn_wmma_f32_16x16x32_f16(false, a, false, b, (short)0, c,
                                                false, false);
}

// ---------------- async global -> LDS (ASYNCcnt-tracked DMA) -----------------
// addrspace(3) pointers are the low 32 bits of the generic LDS address.
__device__ __forceinline__ unsigned lds_addr(const void* p) {
  return (unsigned)(uintptr_t)p;
}
__device__ __forceinline__ void async_ld_b128(unsigned ldsoff, const void* g) {
  asm volatile("global_load_async_to_lds_b128 %0, %1, off"
               :: "v"(ldsoff), "v"(g) : "memory");
}
__device__ __forceinline__ void wait_async_le(int pending) {
  if (pending) asm volatile("s_wait_asynccnt 0x6" ::: "memory");
  else         asm volatile("s_wait_asynccnt 0x0" ::: "memory");
}
__device__ __forceinline__ void wait_async2_le(int pending) {
  if (pending) asm volatile("s_wait_asynccnt 0x2" ::: "memory");
  else         asm volatile("s_wait_asynccnt 0x0" ::: "memory");
}

// ---------------- DPP row-local 16-lane reductions (no LDS traffic) ----------
template <int CTRL>
__device__ __forceinline__ float dpp_f(float x) {
  return __builtin_bit_cast(
      float, __builtin_amdgcn_update_dpp(0, __builtin_bit_cast(int, x), CTRL,
                                         0xF, 0xF, true));
}
__device__ __forceinline__ float row16_max(float x) {
  x = fmaxf(x, dpp_f<0xB1>(x));   // quad_perm xor1
  x = fmaxf(x, dpp_f<0x4E>(x));   // quad_perm xor2
  x = fmaxf(x, dpp_f<0x141>(x));  // row_half_mirror
  x = fmaxf(x, dpp_f<0x140>(x));  // row_mirror
  return x;
}
__device__ __forceinline__ float row16_sum(float x) {
  x += dpp_f<0xB1>(x);
  x += dpp_f<0x4E>(x);
  x += dpp_f<0x141>(x);
  x += dpp_f<0x140>(x);
  return x;
}

// ---------------- Kernel 1: LayerNorm rows of x (C=1024), cast f16 ----------
__global__ __launch_bounds__(256) void ln_rows_kernel(const float* __restrict__ x,
                                                      _Float16* __restrict__ y) {
  int row = blockIdx.x;
  const float* xr = x + (size_t)row * C_;
  int t = threadIdx.x;
  float vals[4], s = 0.f, ss = 0.f;
#pragma unroll
  for (int i = 0; i < 4; ++i) {
    float v = xr[t + 256 * i];
    vals[i] = v; s += v; ss += v * v;
  }
#pragma unroll
  for (int off = 1; off < 32; off <<= 1) {
    s  += __shfl_xor(s, off, 32);
    ss += __shfl_xor(ss, off, 32);
  }
  __shared__ float sh[16];
  int wid = t >> 5;
  if ((t & 31) == 0) { sh[wid] = s; sh[wid + 8] = ss; }
  __syncthreads();
  float S = 0.f, SS = 0.f;
#pragma unroll
  for (int w = 0; w < 8; ++w) { S += sh[w]; SS += sh[w + 8]; }
  float mean = S * (1.0f / C_);
  float var  = SS * (1.0f / C_) - mean * mean;
  float rstd = rsqrtf(var + 1e-6f);
  _Float16* yr = y + (size_t)row * C_;
#pragma unroll
  for (int i = 0; i < 4; ++i)
    yr[t + 256 * i] = (_Float16)((vals[i] - mean) * rstd);
}

// ---------------- Kernel 2: f32 -> f16 convert -------------------------------
__global__ __launch_bounds__(256) void cvt_f16_kernel(const float* __restrict__ in,
                                                      _Float16* __restrict__ out,
                                                      int n) {
  int i = blockIdx.x * 256 + threadIdx.x;
  if (i < n) out[i] = (_Float16)in[i];
}

// ---------------- Kernel 3: 128x256-per-block WMMA GEMM ----------------------
// Block = 8 waves (2x4), each wave a 64x64 tile. Per 32-wide K-step the block
// stages A(128x32) + W(256x32) in LDS via async DMA, double-buffered.
// out[m][c] = sum_k A[m][k] * W[c][k]
// MODE 0: f16 -> [B*H][N][64]; MODE 1: f16 -> [B*H][64][N]; MODE 2: f32+bias
template <int MODE>
__global__ __launch_bounds__(256) void gemm64_kernel(
    const _Float16* __restrict__ A, const _Float16* __restrict__ W,
    _Float16* __restrict__ outh, float* __restrict__ outf,
    const float* __restrict__ bias) {
  __shared__ __align__(16) _Float16 smem[2 * (128 + 256) * LDA];  // 60 KB
  int tid = threadIdx.x;
  int wib = tid >> 5, lane = tid & 31;
  int bm = blockIdx.x >> 2, bn = blockIdx.x & 3;   // (M/128) x (C/256)
  int m0b = bm * 128, n0b = bn * 256;
  int wm = wib >> 2, wn = wib & 3;

  const char* Ag = (const char*)A;
  const char* Wg = (const char*)W;
  _Float16* buf0 = smem;
  _Float16* buf1 = smem + (128 + 256) * LDA;

  int ra = wib * 16 + (lane >> 2);   // this wave's A rows (2 x 8-row groups)
  int rw = wib * 32 + (lane >> 2);   // this wave's W rows (4 x 8-row groups)
  int cb = (lane & 3) * 16;          // byte column within 64B row slice

  auto issue = [&](_Float16* buf, int k0) {
    unsigned bA = lds_addr(buf);
    unsigned bW = bA + 128 * LDA * 2;
#pragma unroll
    for (int t = 0; t < 2; ++t) {
      int r = ra + t * 8;
      async_ld_b128(bA + r * (LDA * 2) + cb,
                    Ag + (size_t)(m0b + r) * (C_ * 2) + k0 * 2 + cb);
    }
#pragma unroll
    for (int t = 0; t < 4; ++t) {
      int r = rw + t * 8;
      async_ld_b128(bW + r * (LDA * 2) + cb,
                    Wg + (size_t)(n0b + r) * (C_ * 2) + k0 * 2 + cb);
    }
  };

  v8f acc[4][4];
  v8f z = {0.f, 0.f, 0.f, 0.f, 0.f, 0.f, 0.f, 0.f};
#pragma unroll
  for (int i = 0; i < 4; ++i)
#pragma unroll
    for (int j = 0; j < 4; ++j) acc[i][j] = z;

  issue(buf0, 0);
  for (int ks = 0; ks < C_ / 32; ++ks) {
    _Float16* cur = (ks & 1) ? buf1 : buf0;
    _Float16* nxt = (ks & 1) ? buf0 : buf1;
    int more = (ks + 1 < C_ / 32);
    if (more) issue(nxt, (ks + 1) * 32);
    wait_async_le(more);   // first 6 (cur) complete; async loads retire in order
    __syncthreads();

    const _Float16* tA = cur;
    const _Float16* tW = cur + 128 * LDA;
    v16h af[4];
#pragma unroll
    for (int i = 0; i < 4; ++i)
      af[i] = load_a(tA + (size_t)(wm * 64 + i * 16) * LDA, LDA, lane);
#pragma unroll
    for (int j = 0; j < 4; ++j) {
      v16h bf = load_b(tW + (size_t)(wn * 64 + j * 16) * LDA, LDA, lane);
#pragma unroll
      for (int i = 0; i < 4; ++i) acc[i][j] = wmma16(af[i], bf, acc[i][j]);
    }
    __syncthreads();   // all waves done reading before buffer is re-filled
  }

  int m0 = m0b + wm * 64, n0 = n0b + wn * 64;
  int col16 = lane & 15, mh = lane >> 4;
#pragma unroll
  for (int i = 0; i < 4; ++i) {
#pragma unroll
    for (int j = 0; j < 4; ++j) {
      int c = n0 + j * 16 + col16;
#pragma unroll
      for (int v = 0; v < 8; ++v) {
        int m = m0 + i * 16 + v + 8 * mh;
        float val = acc[i][j][v];
        if (MODE == 2) {
          outf[(size_t)m * C_ + c] = val + bias[c];
        } else {
          int b = m >> 10, seq = m & (N_ - 1);
          int h = c >> 6, d = c & (DH_ - 1);
          size_t idx = (MODE == 0)
              ? (((size_t)(b * H_ + h) * N_ + seq) * DH_ + d)
              : (((size_t)(b * H_ + h) * DH_ + d) * N_ + seq);
          outh[idx] = (_Float16)val;
        }
      }
    }
  }
}

// ---------------- Kernel 4: per-head LayerNorm over Dh=64 (+scale) ----------
__global__ __launch_bounds__(256) void headln_kernel(_Float16* __restrict__ buf,
                                                     float scale) {
  int gid  = blockIdx.x * 256 + threadIdx.x;
  int wave = gid >> 5;
  int lane = gid & 31;
  _Float16* r = buf + (size_t)wave * DH_;
  float a = (float)r[lane * 2];
  float b = (float)r[lane * 2 + 1];
  float s = a + b, ss = a * a + b * b;
#pragma unroll
  for (int off = 1; off < 32; off <<= 1) {
    s  += __shfl_xor(s, off, 32);
    ss += __shfl_xor(ss, off, 32);
  }
  float mean = s * (1.0f / DH_);
  float var  = ss * (1.0f / DH_) - mean * mean;
  float rstd = rsqrtf(var + 1e-6f) * scale;
  r[lane * 2]     = (_Float16)((a - mean) * rstd);
  r[lane * 2 + 1] = (_Float16)((b - mean) * rstd);
}

// ---------------- Kernel 5: flash attention, 32 query rows per wave ----------
// All 8 waves of a block share (b,h): K(32x64) and Vt(64x32) tiles are staged
// once per block via async DMA (double-buffered), instead of 8x from global.
__global__ __launch_bounds__(256) void attn_kernel(
    const _Float16* __restrict__ Q, const _Float16* __restrict__ K,
    const _Float16* __restrict__ Vt, _Float16* __restrict__ ctx) {
  __shared__ __align__(16) _Float16 sK[2][32 * LDK];   // 9 KB
  __shared__ __align__(16) _Float16 sV[2][64 * LDA];   // 10 KB
  __shared__ __align__(16) _Float16 plds[8 * 32 * 32]; // 16 KB
  int tid  = threadIdx.x;
  int wib  = tid >> 5;
  int lane = tid & 31;
  int wave = blockIdx.x * 8 + wib;
  int bh = wave >> 5;              // shared by all 8 waves of the block
  int mt = wave & 31;
  int b  = bh >> 4, h = bh & 15;
  int m0 = mt * 32;

  const _Float16* Qb  = Q  + (size_t)bh * N_ * DH_ + (size_t)m0 * DH_;
  const _Float16* Kb0 = K  + (size_t)bh * N_ * DH_;
  const _Float16* Vb0 = Vt + (size_t)bh * DH_ * N_;
  _Float16* pl = plds + wib * (32 * 32);

  int rk = wib * 4 + (lane >> 3);  // K stage: 4 rows/wave, 128B rows
  int ck = (lane & 7) * 16;
  int rv = wib * 8 + (lane >> 2);  // V stage: 8 rows/wave, 64B rows
  int cv = (lane & 3) * 16;

  auto issue = [&](int bufsel, int kb) {
    const char* Kg = (const char*)(Kb0 + (size_t)kb * 32 * DH_);
    const char* Vg = (const char*)(Vb0 + kb * 32);
    async_ld_b128(lds_addr(&sK[bufsel][0]) + rk * (LDK * 2) + ck,
                  Kg + (size_t)rk * (DH_ * 2) + ck);
    async_ld_b128(lds_addr(&sV[bufsel][0]) + rv * (LDA * 2) + cv,
                  Vg + (size_t)rv * (N_ * 2) + cv);
  };

  v16h aq[2][2];
#pragma unroll
  for (int mi = 0; mi < 2; ++mi) {
    aq[mi][0] = load_a(Qb + (size_t)mi * 16 * DH_, DH_, lane);
    aq[mi][1] = load_a(Qb + (size_t)mi * 16 * DH_ + 32, DH_, lane);
  }

  v8f z = {0.f, 0.f, 0.f, 0.f, 0.f, 0.f, 0.f, 0.f};
  v8f o[2][4];
  float mrow[2][8], lrow[2][8];
#pragma unroll
  for (int mi = 0; mi < 2; ++mi) {
#pragma unroll
    for (int j = 0; j < 4; ++j) o[mi][j] = z;
#pragma unroll
    for (int v = 0; v < 8; ++v) { mrow[mi][v] = -1e30f; lrow[mi][v] = 0.f; }
  }

  int col = lane & 15, mh = lane >> 4;

  issue(0, 0);
  for (int kb = 0; kb < N_ / 32; ++kb) {
    int cur = kb & 1;
    int more = (kb + 1 < N_ / 32);
    if (more) issue(1 - cur, kb + 1);
    wait_async2_le(more);
    __syncthreads();

    const _Float16* Kh = sK[cur];
    const _Float16* Vh = sV[cur];
    v16h bk0 = load_b(Kh, LDK, lane);                  // keys 0..15, d 0..31
    v16h bk1 = load_b(Kh + 32, LDK, lane);             // keys 0..15, d 32..63
    v16h bk2 = load_b(Kh + 16 * LDK, LDK, lane);       // keys 16..31, d 0..31
    v16h bk3 = load_b(Kh + 16 * LDK + 32, LDK, lane);  // keys 16..31, d 32..63

    v8f s[2][2];
#pragma unroll
    for (int mi = 0; mi < 2; ++mi) {
      s[mi][0] = wmma16(aq[mi][0], bk0, z);
      s[mi][0] = wmma16(aq[mi][1], bk1, s[mi][0]);
      s[mi][1] = wmma16(aq[mi][0], bk2, z);
      s[mi][1] = wmma16(aq[mi][1], bk3, s[mi][1]);
    }

#pragma unroll
    for (int mi = 0; mi < 2; ++mi) {
      float alpha[8];
#pragma unroll
      for (int v = 0; v < 8; ++v) {
        float mx   = row16_max(fmaxf(s[mi][0][v], s[mi][1][v]));
        float mnew = fmaxf(mrow[mi][v], mx);
        float al = __expf(mrow[mi][v] - mnew);
        float p0 = __expf(s[mi][0][v] - mnew);
        float p1 = __expf(s[mi][1][v] - mnew);
        float rs = row16_sum(p0 + p1);
        lrow[mi][v] = lrow[mi][v] * al + rs;
        mrow[mi][v] = mnew;
        alpha[v] = al;
        s[mi][0][v] = p0;
        s[mi][1][v] = p1;
      }
#pragma unroll
      for (int j = 0; j < 4; ++j)
#pragma unroll
        for (int v = 0; v < 8; ++v) o[mi][j][v] *= alpha[v];

#pragma unroll
      for (int v = 0; v < 8; ++v) {
        int row = v + 8 * mh;
        pl[(mi * 16 + row) * 32 + col]      = (_Float16)s[mi][0][v];
        pl[(mi * 16 + row) * 32 + 16 + col] = (_Float16)s[mi][1][v];
      }
    }

    v16h ap0 = load_a(pl, 32, lane);
    v16h ap1 = load_a(pl + 16 * 32, 32, lane);

#pragma unroll
    for (int j = 0; j < 4; ++j) {
      v16h bv = load_b(Vh + (size_t)(j * 16) * LDA, LDA, lane);
      o[0][j] = wmma16(ap0, bv, o[0][j]);
      o[1][j] = wmma16(ap1, bv, o[1][j]);
    }
    __syncthreads();
  }

#pragma unroll
  for (int mi = 0; mi < 2; ++mi) {
    float inv[8];
#pragma unroll
    for (int v = 0; v < 8; ++v) inv[v] = 1.0f / lrow[mi][v];
#pragma unroll
    for (int j = 0; j < 4; ++j)
#pragma unroll
      for (int v = 0; v < 8; ++v) {
        int row = v + 8 * mh;
        size_t idx = ((size_t)(b * N_ + m0 + mi * 16 + row)) * C_ +
                     h * DH_ + j * 16 + col;
        ctx[idx] = (_Float16)(o[mi][j][v] * inv[v]);
      }
  }
}

// ---------------- Host launcher ----------------------------------------------
extern "C" void kernel_launch(void* const* d_in, const int* in_sizes, int n_in,
                              void* d_out, int out_size, void* d_ws, size_t ws_size,
                              hipStream_t stream) {
  (void)in_sizes; (void)n_in; (void)out_size; (void)ws_size;
  const float* x  = (const float*)d_in[0];
  const float* Wq = (const float*)d_in[1];
  const float* Wk = (const float*)d_in[2];
  const float* Wv = (const float*)d_in[3];
  const float* Wp = (const float*)d_in[4];
  const float* bp = (const float*)d_in[5];
  float* out = (float*)d_out;

  char* ws = (char*)d_ws;
  size_t off = 0;
  auto alloc = [&](size_t bytes) -> void* {
    void* p = ws + off;
    off += (bytes + 255) & ~(size_t)255;
    return p;
  };
  const size_t big = (size_t)M_ * C_ * sizeof(_Float16);   // 32 MB
  _Float16* yh  = (_Float16*)alloc(big);
  _Float16* qh  = (_Float16*)alloc(big);
  _Float16* kh  = (_Float16*)alloc(big);
  _Float16* vt  = (_Float16*)alloc(big);
  _Float16* wqh = (_Float16*)alloc((size_t)C_ * C_ * 2);
  _Float16* wkh = (_Float16*)alloc((size_t)C_ * C_ * 2);
  _Float16* wvh = (_Float16*)alloc((size_t)C_ * C_ * 2);
  _Float16* wph = (_Float16*)alloc((size_t)C_ * C_ * 2);
  _Float16* ctx = yh;  // y dead after the three projections; reuse

  int nW = C_ * C_;
  ln_rows_kernel<<<M_, 256, 0, stream>>>(x, yh);
  cvt_f16_kernel<<<nW / 256, 256, 0, stream>>>(Wq, wqh, nW);
  cvt_f16_kernel<<<nW / 256, 256, 0, stream>>>(Wk, wkh, nW);
  cvt_f16_kernel<<<nW / 256, 256, 0, stream>>>(Wv, wvh, nW);
  cvt_f16_kernel<<<nW / 256, 256, 0, stream>>>(Wp, wph, nW);

  // 512 blocks: (M/128) x (C/256), 8 waves each
  gemm64_kernel<0><<<512, 256, 0, stream>>>(yh, wqh, qh, nullptr, nullptr);
  gemm64_kernel<0><<<512, 256, 0, stream>>>(yh, wkh, kh, nullptr, nullptr);
  gemm64_kernel<1><<<512, 256, 0, stream>>>(yh, wvh, vt, nullptr, nullptr);

  headln_kernel<<<32768, 256, 0, stream>>>(qh, 0.125f);
  headln_kernel<<<32768, 256, 0, stream>>>(kh, 1.0f);

  // flash attention: 8192 waves (BH * N/32), 8 per block, shared K/V staging
  attn_kernel<<<1024, 256, 0, stream>>>(qh, kh, vt, ctx);

  gemm64_kernel<2><<<512, 256, 0, stream>>>(ctx, wph, nullptr, out, bp);
}
